// KernelNN_11536282157493
// MI455X (gfx1250) — matrix operand order repaired and checked by reading, not compiled, not run on hardware
//
#include <hip/hip_runtime.h>

typedef __attribute__((ext_vector_type(16))) _Float16 v16h;
typedef __attribute__((ext_vector_type(8)))  _Float16 v8h;
typedef __attribute__((ext_vector_type(8)))  float    v8f;

// LDS K-strides (in halves). Padded to break bank conflicts (272B stride -> 4-bank step).
#define LDK1 40
#define LDK  136
#define W_K1 (128 * LDK1)              // 5120 halves  : k1^T  [N=128][K=40]  (K 6 -> pad 40)
#define W_K2 (128 * LDK)               // 17408 halves : k2^T  [N=128][K=136] (K 128 -> pad 136)
#define W_K3 (256 * LDK)               // 34816 halves : k3^T  [N=256][K=136]
#define W_TOT (W_K1 + W_K2 + W_K3)     // 57344 halves = 114,688 B

__device__ __forceinline__ v8f wmma16x16x32(v16h a, v16h b, v8f c) {
  return __builtin_amdgcn_wmma_f32_16x16x32_f16(false, a, false, b, (short)0, c,
                                                false, false);
}

// B fragment (16x16 N-tile of a KxN matrix stored transposed [N][ldk] in LDS).
// Lane l: column n = n0 + (l&15); K values k0+16*(l>=16) .. +15, contiguous.
__device__ __forceinline__ v16h load_bfrag(const _Float16* wp, int ldk, int k0,
                                           int n0, int lane) {
  int n  = n0 + (lane & 15);
  int kb = k0 + ((lane >> 4) << 4);
  const _Float16* p = wp + n * ldk + kb;
  v8h lo = *(const v8h*)(p);
  v8h hi = *(const v8h*)(p + 8);
  v16h b;
#pragma unroll
  for (int i = 0; i < 8; ++i) { b[i] = lo[i]; b[i + 8] = hi[i]; }
  return b;
}

// A fragment from per-wave row-major f16 staging [16][LDK].
// Lane l: row m = l&15; K values kb..kb+15 contiguous, kb = k0 + 16*(l>=16).
__device__ __forceinline__ v16h load_afrag(const _Float16* stp, int k0, int lane) {
  int m  = lane & 15;
  int kb = k0 + ((lane >> 4) << 4);
  const _Float16* p = stp + m * LDK + kb;
  v8h lo = *(const v8h*)(p);
  v8h hi = *(const v8h*)(p + 8);
  v16h a;
#pragma unroll
  for (int i = 0; i < 8; ++i) { a[i] = lo[i]; a[i + 8] = hi[i]; }
  return a;
}

// ---- convert kernel-MLP weights to f16, transposed + K-padded --------------
__global__ void prep_weights(const float* __restrict__ k1, const float* __restrict__ k2,
                             const float* __restrict__ k3, _Float16* __restrict__ wp) {
  int t = blockIdx.x * blockDim.x + threadIdx.x;
  if (t < W_K1) {
    int n = t / LDK1, k = t % LDK1;
    wp[t] = (k < 6) ? (_Float16)k1[k * 128 + n] : (_Float16)0.f;
  } else if (t < W_K1 + W_K2) {
    int u = t - W_K1; int n = u / LDK, k = u % LDK;
    wp[t] = (k < 128) ? (_Float16)k2[k * 128 + n] : (_Float16)0.f;
  } else if (t < W_TOT) {
    int u = t - (W_K1 + W_K2); int n = u / LDK, k = u % LDK;
    wp[t] = (k < 128) ? (_Float16)k3[k * 256 + n] : (_Float16)0.f;
  }
}

// ---- fused edge-kernel MLP: Wk[e] = (relu(relu(attr@k1+b1)@k2+b2))@k3+b3 ----
// 8 waves/block, 16 edges/wave. All GEMMs via v_wmma_f32_16x16x32_f16.
// Weights staged into LDS with GLOBAL_LOAD_ASYNC_TO_LDS_B128 (ASYNCcnt path).
__global__ __launch_bounds__(256) void mlp_wk(
    const _Float16* __restrict__ wpk, const float* __restrict__ ea,
    const float* __restrict__ b1, const float* __restrict__ b2,
    const float* __restrict__ b3, float* __restrict__ Wk, int E) {
  __shared__ __align__(16) _Float16 lw[W_TOT];        // 114,688 B weights
  __shared__ __align__(16) _Float16 st[8 * 16 * LDK]; //  34,816 B staging

  { // async global->LDS weight copy, no VGPR staging (CDNA5 async path).
    // Generic LDS pointer is {SHARED_BASE, lds_offset[31:0]} -> low 32 bits
    // are the addrspace(3) byte offset the async instruction consumes.
    unsigned ldsbase = (unsigned)(unsigned long long)(void*)&lw[0];
    unsigned long long gbase = (unsigned long long)(const void*)wpk;
    for (int t = threadIdx.x; t < W_TOT / 8; t += 256) {
      unsigned      laddr = ldsbase + (unsigned)t * 16u;
      unsigned long long gaddr = gbase + (unsigned long long)t * 16ull;
      asm volatile("global_load_async_to_lds_b128 %0, %1, off"
                   :: "v"(laddr), "v"(gaddr) : "memory");
    }
    asm volatile("s_wait_asynccnt 0x0" ::: "memory");
  }
  __syncthreads();

  const int lane = threadIdx.x & 31;
  const int wid  = threadIdx.x >> 5;
  const int e0   = (blockIdx.x * 8 + wid) * 16;   // wave-uniform
  if (e0 >= E) return;

  _Float16* myst = st + wid * (16 * LDK);
  const _Float16* k1p = lw;
  const _Float16* k2p = lw + W_K1;
  const _Float16* k3p = lw + W_K1 + W_K2;

  const int nn = lane & 15;
  const int mh = (lane >> 4) << 3;   // 0 or 8: D-matrix row offset for this half-wave

  // A fragment straight from global edge_attr (K=6, zero padded to 32).
  // Upper half-wave covers K=16..31 -> entirely zero.
  v16h a0;
#pragma unroll
  for (int i = 0; i < 16; ++i) a0[i] = (_Float16)0.f;
  if (lane < 16) {
    int e = e0 + nn; if (e >= E) e = E - 1;
    const float* ap = ea + (size_t)e * 6;
#pragma unroll
    for (int i = 0; i < 6; ++i) a0[i] = (_Float16)ap[i];
  }

  // -------- h1 = relu(attr @ k1 + b1)  [16 x 128] -> staging (f16)
#pragma unroll
  for (int nt = 0; nt < 8; ++nt) {
    v8f c = {};
    c = wmma16x16x32(a0, load_bfrag(k1p, LDK1, 0, nt * 16, lane), c);
    float bb = b1[nt * 16 + nn];
#pragma unroll
    for (int v = 0; v < 8; ++v) {
      float val = c[v] + bb;
      val = val > 0.f ? val : 0.f;
      myst[(v + mh) * LDK + nt * 16 + nn] = (_Float16)val;
    }
  }

  // -------- h2 = relu(h1 @ k2 + b2): keep 8 accumulators in VGPRs, then
  // overwrite staging (all h1 reads complete first; LDS ops are in-order).
  v8f hacc[8];
#pragma unroll
  for (int nt = 0; nt < 8; ++nt) { v8f z = {}; hacc[nt] = z; }
#pragma unroll
  for (int ks = 0; ks < 4; ++ks) {
    v16h a = load_afrag(myst, ks * 32, lane);
#pragma unroll
    for (int nt = 0; nt < 8; ++nt)
      hacc[nt] = wmma16x16x32(a, load_bfrag(k2p, LDK, ks * 32, nt * 16, lane), hacc[nt]);
  }
#pragma unroll
  for (int nt = 0; nt < 8; ++nt) {
    float bb = b2[nt * 16 + nn];
#pragma unroll
    for (int v = 0; v < 8; ++v) {
      float val = hacc[nt][v] + bb;
      val = val > 0.f ? val : 0.f;
      myst[(v + mh) * LDK + nt * 16 + nn] = (_Float16)val;
    }
  }

  // -------- Wk = h2 @ k3 + b3  [16 x 256] -> global f32
  v16h af0 = load_afrag(myst, 0,  lane);
  v16h af1 = load_afrag(myst, 32, lane);
  v16h af2 = load_afrag(myst, 64, lane);
  v16h af3 = load_afrag(myst, 96, lane);
#pragma unroll
  for (int nt = 0; nt < 16; ++nt) {
    v8f c = {};
    c = wmma16x16x32(af0, load_bfrag(k3p, LDK, 0,  nt * 16, lane), c);
    c = wmma16x16x32(af1, load_bfrag(k3p, LDK, 32, nt * 16, lane), c);
    c = wmma16x16x32(af2, load_bfrag(k3p, LDK, 64, nt * 16, lane), c);
    c = wmma16x16x32(af3, load_bfrag(k3p, LDK, 96, nt * 16, lane), c);
    float bb = b3[nt * 16 + nn];
#pragma unroll
    for (int v = 0; v < 8; ++v) {
      int e = e0 + v + mh;
      if (e < E) Wk[(size_t)e * 256 + nt * 16 + nn] = c[v] + bb;
    }
  }
}

// ---- helpers ---------------------------------------------------------------
__global__ void zero_kernel(float* p, int n) {
  int t = blockIdx.x * blockDim.x + threadIdx.x;
  if (t < n) p[t] = 0.f;
}

__global__ void deg_kernel(const int* __restrict__ ei, float* __restrict__ deg, int E) {
  int e = blockIdx.x * blockDim.x + threadIdx.x;
  if (e < E) atomicAdd(&deg[ei[E + e]], 1.f);
}

__global__ void feat_init(const float* __restrict__ x, const float* __restrict__ w,
                          const float* __restrict__ b, float* __restrict__ feat, int N) {
  int t = blockIdx.x * blockDim.x + threadIdx.x;
  if (t < N * 16) {
    int n = t >> 4, o = t & 15;
    feat[t] = fmaf(x[n], w[o], b[o]);
  }
}

// 16 lanes per edge: lane j = output channel. Wk reads fully coalesced,
// feat[src] broadcast within the group, scatter via f32 atomics (contiguous 64B).
__global__ __launch_bounds__(256) void msg_kernel(
    const int* __restrict__ ei, const float* __restrict__ Wk,
    const float* __restrict__ feat, float* __restrict__ agg, int E) {
  int t = blockIdx.x * blockDim.x + threadIdx.x;
  int e = t >> 4, j = t & 15;
  if (e >= E) return;
  int s = ei[e], d = ei[E + e];
  const float* wrow = Wk + (size_t)e * 256 + j;
  const float* f = feat + (size_t)s * 16;
  float acc = 0.f;
#pragma unroll
  for (int i = 0; i < 16; ++i) acc = fmaf(f[i], wrow[i * 16], acc);
  atomicAdd(&agg[(size_t)d * 16 + j], acc);
}

__global__ __launch_bounds__(256) void node_update(
    const float* __restrict__ agg, const float* __restrict__ deg,
    const float* __restrict__ fin, const float* __restrict__ rootw,
    const float* __restrict__ convb, float* __restrict__ fout, int N) {
  __shared__ float rw[256];
  __shared__ float cb[16];
  rw[threadIdx.x] = rootw[threadIdx.x];
  if (threadIdx.x < 16) cb[threadIdx.x] = convb[threadIdx.x];
  __syncthreads();
  int n = blockIdx.x * blockDim.x + threadIdx.x;
  if (n >= N) return;
  float f[16];
#pragma unroll
  for (int i = 0; i < 16; ++i) f[i] = fin[(size_t)n * 16 + i];
  float dg = deg[n];
  if (dg < 1.f) dg = 1.f;
  float inv = 1.f / dg;
#pragma unroll
  for (int o = 0; o < 16; ++o) {
    float v = fmaf(agg[(size_t)n * 16 + o], inv, cb[o]);
#pragma unroll
    for (int i = 0; i < 16; ++i) v = fmaf(f[i], rw[i * 16 + o], v);
    fout[(size_t)n * 16 + o] = v > 0.f ? v : 0.f;
  }
}

__global__ void out_kernel(const float* __restrict__ feat, const float* __restrict__ w,
                           const float* __restrict__ b, float* __restrict__ out, int N) {
  int n = blockIdx.x * blockDim.x + threadIdx.x;
  if (n >= N) return;
  float acc = b[0];
#pragma unroll
  for (int o = 0; o < 16; ++o) acc = fmaf(feat[(size_t)n * 16 + o], w[o], acc);
  out[n] = acc;
}

extern "C" void kernel_launch(void* const* d_in, const int* in_sizes, int n_in,
                              void* d_out, int out_size, void* d_ws, size_t ws_size,
                              hipStream_t stream) {
  const float* x     = (const float*)d_in[0];
  const int*   ei    = (const int*)d_in[1];   // edge_index [2,E]
  const float* ea    = (const float*)d_in[2]; // edge_attr [E,6]
  const float* fc1w  = (const float*)d_in[3];
  const float* fc1b  = (const float*)d_in[4];
  const float* k1w   = (const float*)d_in[5];
  const float* k1b   = (const float*)d_in[6];
  const float* k2w   = (const float*)d_in[7];
  const float* k2b   = (const float*)d_in[8];
  const float* k3w   = (const float*)d_in[9];
  const float* k3b   = (const float*)d_in[10];
  const float* rootw = (const float*)d_in[11];
  const float* convb = (const float*)d_in[12];
  const float* fc2w  = (const float*)d_in[13];
  const float* fc2b  = (const float*)d_in[14];
  float* out = (float*)d_out;

  const int N = in_sizes[0];
  const int E = in_sizes[2] / 6;

  // workspace carve-out (256B aligned)
  char* ws = (char*)d_ws;
  size_t off = 0;
  auto take = [&](size_t bytes) -> char* {
    char* p = ws + off;
    off = (off + bytes + 255) & ~(size_t)255;
    return p;
  };
  float*    Wk  = (float*)take((size_t)E * 256 * sizeof(float)); // ~410 MB
  _Float16* wpk = (_Float16*)take((size_t)W_TOT * sizeof(_Float16));
  float*    deg = (float*)take((size_t)N * sizeof(float));
  float*    fA  = (float*)take((size_t)N * 16 * sizeof(float));
  float*    fB  = (float*)take((size_t)N * 16 * sizeof(float));
  float*    agg = (float*)take((size_t)N * 16 * sizeof(float));

  prep_weights<<<(W_TOT + 255) / 256, 256, 0, stream>>>(k1w, k2w, k3w, wpk);

  zero_kernel<<<(N + 255) / 256, 256, 0, stream>>>(deg, N);
  deg_kernel<<<(E + 255) / 256, 256, 0, stream>>>(ei, deg, E);

  mlp_wk<<<(E + 127) / 128, 256, 0, stream>>>(wpk, ea, k1b, k2b, k3b, Wk, E);

  feat_init<<<(N * 16 + 255) / 256, 256, 0, stream>>>(x, fc1w, fc1b, fA, N);

  float* fin = fA;
  float* fout = fB;
  for (int d = 0; d < 4; ++d) {
    zero_kernel<<<(N * 16 + 255) / 256, 256, 0, stream>>>(agg, N * 16);
    msg_kernel<<<(int)(((size_t)E * 16 + 255) / 256), 256, 0, stream>>>(ei, Wk, fin, agg, E);
    node_update<<<(N + 255) / 256, 256, 0, stream>>>(agg, deg, fin, rootw, convb, fout, N);
    float* tmp = fin; fin = fout; fout = tmp;
  }

  out_kernel<<<(N + 255) / 256, 256, 0, stream>>>(fin, fc2w, fc2b, out, N);
}
